// FastWeightMemory_32366873543105
// MI455X (gfx1250) — matrix operand order repaired and checked by reading, not compiled, not run on hardware
//
#include <hip/hip_runtime.h>

// ---------------------------------------------------------------------------
// FastWeightMemory for MI455X (gfx1250, wave32)
//  - bf16 WMMA (v_wmma_f32_16x16x32_bf16) for the large projections (mem-bound)
//  - fp32 WMMA (v_wmma_f32_16x16x4_f32) for the chunked linear-attention scan
//  - global_load_async_to_lds_b128 (ASYNCcnt) ping-pong prefetch in the scan
//  - ping-pong fp32 fw state in LDS; branchless hoisted decay factors
//  - double-buffered LDS pipeline in the GEMM (1 barrier / k-step)
// ---------------------------------------------------------------------------

typedef __bf16 bf16_t;
typedef __attribute__((ext_vector_type(16))) __bf16 v16bf;
typedef __attribute__((ext_vector_type(8)))  float  v8f;
typedef __attribute__((ext_vector_type(4)))  float  v4f;
typedef __attribute__((ext_vector_type(2)))  float  v2f;

#define NHEAD   8
#define DKDIM   64
#define DVDIM   64
#define DMODEL  768
#define SEQLEN  4096
#define NBATCH  4
#define DECAYF  0.95f
#define WSTRF   0.1f

__device__ __forceinline__ unsigned lds_off(const void* p) {
  // generic LDS pointer: low 32 bits are the LDS byte offset
  return (unsigned)(uintptr_t)p;
}

__device__ __forceinline__ void async_b128(unsigned ldsoff, const void* gaddr) {
  asm volatile("global_load_async_to_lds_b128 %0, %1, off"
               :: "v"(ldsoff), "v"((unsigned long long)(uintptr_t)gaddr)
               : "memory");
}

template <int N>
__device__ __forceinline__ void wait_async() {
  asm volatile("s_wait_asynccnt %0" :: "i"(N) : "memory");
}

// ======================= bf16 tiled GEMM: C = A @ W^T (+res) ================
// A: M x K row-major fp32, W: N x K row-major fp32 (torch Linear weight)
// M % 128 == 0, N % 64 == 0, K % 32 == 0 (holds for all calls here)
#define BM 128
#define BN 64
#define BK 32
#define LDAS (BK + 8)
#define LDWS (BK + 8)
#define LDCS (BN + 4)

__device__ __forceinline__ v16bf load_a_frag(const bf16_t* base, int ld,
                                             int mbase, int lane) {
  // 16x32 bf16 A tile: lane&15 = row, lane>>4 selects K-interleave half
  const int row = mbase + (lane & 15);
  const int hs  = (lane >> 4) << 3;         // 0 or 8
  const bf16_t* p = base + row * ld;
  v16bf a;
#pragma unroll
  for (int j = 0; j < 8; ++j) a[j] = p[hs + j];          // K = hs+0..7
#pragma unroll
  for (int j = 0; j < 8; ++j) a[8 + j] = p[16 + hs + j]; // K = 16+hs+0..7
  return a;
}

__device__ __forceinline__ v16bf load_b_frag(const bf16_t* base, int ld,
                                             int nbase, int lane) {
  // 32x16 bf16 B tile: lane&15 = col N, lane>>4 selects K-interleave half
  const int col = nbase + (lane & 15);
  const int ks  = (lane >> 4) << 3;         // 0 or 8
  const bf16_t* p = base + col * ld + ks;
  v16bf b;
#pragma unroll
  for (int j = 0; j < 8; ++j) { b[j] = p[j]; b[8 + j] = p[16 + j]; }
  return b;
}

__device__ __forceinline__ void cvt_store4(bf16_t* sp, v4f t) {
  sp[0] = (bf16_t)t[0]; sp[1] = (bf16_t)t[1];
  sp[2] = (bf16_t)t[2]; sp[3] = (bf16_t)t[3];
}

template <bool HAS_RES>
__global__ __launch_bounds__(256)
void gemm_bf16_xwT(const float* __restrict__ A, const float* __restrict__ W,
                   const float* __restrict__ residual,
                   float* __restrict__ C, int M, int N, int K) {
  __shared__ bf16_t As[2][BM * LDAS];
  __shared__ bf16_t Ws[2][BN * LDWS];
  __shared__ float  Cs[BM * LDCS];

  const int tid  = threadIdx.x;
  const int lane = tid & 31;
  const int wave = tid >> 5;     // 8 waves
  const int wm   = wave & 3;     // 4 along M (32 rows each)
  const int wn   = wave >> 2;    // 2 along N (32 cols each)
  const int m0   = blockIdx.y * BM;
  const int n0   = blockIdx.x * BN;

  v8f acc[2][2];
#pragma unroll
  for (int i = 0; i < 2; ++i)
#pragma unroll
    for (int j = 0; j < 2; ++j) acc[i][j] = (v8f){};

  // per-thread staging coordinates (fixed across k-loop)
  const int arow = tid >> 1, akc = (tid & 1) * 16;   // A: 16 floats/thread
  const int wrow = tid >> 2, wkc = (tid & 3) * 8;    // W: 8 floats/thread
  const float* gpA = A + (size_t)(m0 + arow) * K + akc;
  const float* gpW = W + (size_t)(n0 + wrow) * K + wkc;

  // ---- prologue: stage k-tile 0 into buffer 0 ----
  {
    bf16_t* spA = As[0] + arow * LDAS + akc;
    bf16_t* spW = Ws[0] + wrow * LDWS + wkc;
#pragma unroll
    for (int i = 0; i < 16; i += 4) cvt_store4(spA + i, *(const v4f*)(gpA + i));
#pragma unroll
    for (int i = 0; i < 8; i += 4)  cvt_store4(spW + i, *(const v4f*)(gpW + i));
  }
  __syncthreads();

  const int NK = K / BK;
  for (int kt = 0; kt < NK; ++kt) {
    const int p = kt & 1;
    const bool more = (kt + 1 < NK);

    // ---- issue next tile's global loads into registers (overlaps WMMA) ----
    v4f ar[4], wr[2];
    if (more) {
      const float* ga = gpA + (size_t)(kt + 1) * BK;
      const float* gw = gpW + (size_t)(kt + 1) * BK;
#pragma unroll
      for (int i = 0; i < 4; ++i) ar[i] = *(const v4f*)(ga + 4 * i);
#pragma unroll
      for (int i = 0; i < 2; ++i) wr[i] = *(const v4f*)(gw + 4 * i);
      if (kt + 2 < NK) {                      // global_prefetch_b8, 2 ahead
        __builtin_prefetch(ga + BK, 0, 0);
        __builtin_prefetch(gw + BK, 0, 0);
      }
    }

    // ---- compute on buffer p ----
    v16bf af0 = load_a_frag(As[p], LDAS, wm * 32 + 0,  lane);
    v16bf af1 = load_a_frag(As[p], LDAS, wm * 32 + 16, lane);
    v16bf bf0 = load_b_frag(Ws[p], LDWS, wn * 32 + 0,  lane);
    v16bf bf1 = load_b_frag(Ws[p], LDWS, wn * 32 + 16, lane);

    acc[0][0] = __builtin_amdgcn_wmma_f32_16x16x32_bf16(false, af0, false, bf0,
                                                        (short)0, acc[0][0], false, false);
    acc[0][1] = __builtin_amdgcn_wmma_f32_16x16x32_bf16(false, af0, false, bf1,
                                                        (short)0, acc[0][1], false, false);
    acc[1][0] = __builtin_amdgcn_wmma_f32_16x16x32_bf16(false, af1, false, bf0,
                                                        (short)0, acc[1][0], false, false);
    acc[1][1] = __builtin_amdgcn_wmma_f32_16x16x32_bf16(false, af1, false, bf1,
                                                        (short)0, acc[1][1], false, false);

    // ---- convert + store next tile into the other buffer ----
    if (more) {
      bf16_t* spA = As[p ^ 1] + arow * LDAS + akc;
      bf16_t* spW = Ws[p ^ 1] + wrow * LDWS + wkc;
#pragma unroll
      for (int i = 0; i < 4; ++i) cvt_store4(spA + 4 * i, ar[i]);
#pragma unroll
      for (int i = 0; i < 2; ++i) cvt_store4(spW + 4 * i, wr[i]);
    }
    __syncthreads();
  }

  // ---- epilogue: stage C tile in LDS, then coalesced b128 stores ----
  const int crow = (lane >> 4) << 3;
  const int ccol = lane & 15;
#pragma unroll
  for (int mi = 0; mi < 2; ++mi)
#pragma unroll
    for (int ni = 0; ni < 2; ++ni)
#pragma unroll
      for (int r = 0; r < 8; ++r)
        Cs[(wm * 32 + mi * 16 + r + crow) * LDCS + wn * 32 + ni * 16 + ccol] =
            acc[mi][ni][r];
  __syncthreads();

  // 128x64 floats = 2048 float4, 8 per thread, consecutive lanes -> consecutive
#pragma unroll
  for (int i = 0; i < 8; ++i) {
    const int idx = tid + 256 * i;      // float4 index
    const int row = idx >> 4;
    const int c4  = (idx & 15) * 4;
    v4f val = *(const v4f*)&Cs[row * LDCS + c4];
    const size_t goff = (size_t)(m0 + row) * N + n0 + c4;
    if constexpr (HAS_RES) {
      v4f res = *(const v4f*)(residual + goff);
      val[0] += res[0]; val[1] += res[1]; val[2] += res[2]; val[3] += res[3];
    }
    *(v4f*)(C + goff) = val;
  }
}

// ======================= gate: sigmoid(x[0] @ w_gate^T) =====================
__global__ __launch_bounds__(256)
void gate_kernel(const float* __restrict__ x, const float* __restrict__ wg,
                 float* __restrict__ g0, int S, int Dm) {
  const int task = blockIdx.x * 8 + (threadIdx.x >> 5);   // one wave per (s,h)
  const int lane = threadIdx.x & 31;
  const int s = task >> 3, h = task & 7;
  if (s >= S) return;
  const float* xp = x  + (size_t)s * Dm;   // batch 0 only
  const float* wp = wg + (size_t)h * Dm;
  float acc = 0.f;
  for (int d = lane; d < Dm; d += 32) acc += xp[d] * wp[d];
#pragma unroll
  for (int off = 16; off; off >>= 1) acc += __shfl_xor(acc, off, 32);
  if (lane == 0) g0[s * NHEAD + h] = 1.f / (1.f + __expf(-acc));
}

// ======================= chunked decayed linear-attention scan ==============
// grid = (NH, B), 256 threads = 8 waves. fw state ping-pongs between two LDS
// buffers (read fwR, write fwW). All decay powers are hoisted into registers
// before the serial chunk loop; the causal mask is branchless.
#define CHK   32
#define LDQ   (DKDIM + 4)   // 272B rows: 16B-aligned for async b128, bank-shift 4
#define LDV   (DVDIM + 4)
#define LDSP  (CHK + 2)     // 8B-aligned rows for v2f A-frag loads
#define LDFW  (DVDIM + 2)
#define LDR   (DVDIM + 4)

__global__ __launch_bounds__(256)
void fwm_scan(const float* __restrict__ q,   // (B,S,NH,DK)
              const float* __restrict__ k0,  // (S,NH,DK)
              const float* __restrict__ v0,  // (S,NH,DV)
              const float* __restrict__ g0,  // (S,NH)
              float* __restrict__ retr,      // (B,S,NH,DV)
              int S) {
  const int h    = blockIdx.x;
  const int b    = blockIdx.y;
  const int tid  = threadIdx.x;
  const int lane = tid & 31;
  const int wave = tid >> 5;            // 8 waves

  __shared__ float fw0[DKDIM * LDFW];   // ping-pong recurrent state
  __shared__ float fw1[DKDIM * LDFW];
  __shared__ float Qs[2][CHK * LDQ];    // ping-pong chunk buffers
  __shared__ float Ks[2][CHK * LDQ];
  __shared__ float Vs[2][CHK * LDV];
  __shared__ float Sp[CHK * LDSP];      // masked intra-chunk scores
  __shared__ float Rs[CHK * LDR];       // output staging (coalesced stores)
  __shared__ float gs[CHK];
  __shared__ float wsc[CHK];            // 0.1*g[s]*decay^(C-1-s)
  __shared__ float dpow[CHK + 1];       // decay^i

  for (int i = tid; i < DKDIM * LDFW; i += 256) fw0[i] = 0.f;
  if (tid <= CHK) dpow[tid] = __powf(DECAYF, (float)tid);
  __syncthreads();   // init visible before any cross-thread read

  const int l15  = lane & 15;
  const int hi16 = lane >> 4;
  const int crow = hi16 << 3;   // C-frag row offset (+0 / +8)
  const int k2   = hi16 << 1;   // f32 frag K offset (+0 / +2)

  // ---- hoist ALL loop-invariant decay powers into registers ----
  float sfac[8];   // S phase: 0.1*decay^(t-1-s), 0 if s >= t (per wave/lane/r)
  float dfac[8];   // R phase: decay^t_local
  {
    const int tiS = (wave & 3) >> 1;          // S-phase tile mapping (waves 0-3)
    const int sS  = (wave & 1) * 16 + l15;
    const int tiR = wave >> 2;                // R-phase tile mapping
#pragma unroll
    for (int r = 0; r < 8; ++r) {
      const int tS = tiS * 16 + r + crow;
      const int d  = tS - 1 - sS;
      sfac[r] = (d >= 0) ? dpow[d] * WSTRF : 0.f;   // clamp-free: d>=0 checked
      dfac[r] = dpow[tiR * 16 + r + crow];
    }
  }
  const float dC = dpow[CHK];                       // decay^CHK
  const float dw = (tid < CHK) ? dpow[CHK - 1 - tid] : 0.f;

  // staging: 8 floats/thread -> row = tid>>3 (0..31), col = (tid&7)*8
  const int srow = tid >> 3;
  const int scol = (tid & 7) * 8;

  const float* qbase = q  + (((size_t)b * S + srow) * NHEAD + h) * DKDIM + scol;
  const float* kbase = k0 + (((size_t)srow * NHEAD + h)) * DKDIM + scol;
  const float* vbase = v0 + (((size_t)srow * NHEAD + h)) * DVDIM + scol;
  const size_t cstep = (size_t)CHK * NHEAD * DKDIM;   // global stride per chunk

  const unsigned qo0 = lds_off(&Qs[0][srow * LDQ + scol]);
  const unsigned ko0 = lds_off(&Ks[0][srow * LDQ + scol]);
  const unsigned vo0 = lds_off(&Vs[0][srow * LDV + scol]);
  const unsigned qo1 = lds_off(&Qs[1][srow * LDQ + scol]);
  const unsigned ko1 = lds_off(&Ks[1][srow * LDQ + scol]);
  const unsigned vo1 = lds_off(&Vs[1][srow * LDV + scol]);

  // ---- prologue: async-stage chunk 0 into buffer 0 (6 ops/wave) ----
#pragma unroll
  for (int i = 0; i < 2; ++i) {
    async_b128(qo0 + 16 * i, qbase + 4 * i);
    async_b128(ko0 + 16 * i, kbase + 4 * i);
    async_b128(vo0 + 16 * i, vbase + 4 * i);
  }

  const int NC = S / CHK;
  for (int ci = 0; ci < NC; ++ci) {
    const int p  = ci & 1;
    const int c0 = ci * CHK;
    float* fwR = p ? fw1 : fw0;       // state entering this chunk
    float* fwW = p ? fw0 : fw1;       // state leaving this chunk

    // ---- issue async prefetch of chunk ci+1 into the other buffer ----
    if (ci + 1 < NC) {
      const size_t off = (size_t)(ci + 1) * cstep;
      const unsigned qo = p ? qo0 : qo1;
      const unsigned ko = p ? ko0 : ko1;
      const unsigned vo = p ? vo0 : vo1;
#pragma unroll
      for (int i = 0; i < 2; ++i) {
        async_b128(qo + 16 * i, qbase + off + 4 * i);
        async_b128(ko + 16 * i, kbase + off + 4 * i);
        async_b128(vo + 16 * i, vbase + off + 4 * i);
      }
    }
    // gate scalars for this chunk (dw hoisted; one global load / thread)
    if (tid < CHK) {
      float g = g0[(size_t)(c0 + tid) * NHEAD + h];
      gs[tid]  = g;
      wsc[tid] = WSTRF * g * dw;
    }
    // wait only for the CURRENT chunk's 6 transfers (in-order completion)
    if (ci + 1 < NC) wait_async<6>(); else wait_async<0>();
    __syncthreads();

    const float* Q  = Qs[p];
    const float* Kc = Ks[p];
    const float* Vc = Vs[p];

    // ---- S = Q @ K^T (32x32); waves 0-3 -> tile (ti,si); mask -> Sp ----
    if (wave < 4) {
      const int ti = wave >> 1, si = wave & 1;
      const int qrow = (ti * 16 + l15) * LDQ;
      const int krow = (si * 16 + l15) * LDQ;
      v8f sacc = (v8f){};
#pragma unroll
      for (int kk = 0; kk < DKDIM; kk += 4) {
        v2f a  = *(const v2f*)&Q[qrow + kk + k2];
        v2f bb = *(const v2f*)&Kc[krow + kk + k2];   // B = K^T: col=s, k=dk
        sacc = __builtin_amdgcn_wmma_f32_16x16x4_f32(false, a, false, bb,
                                                     (short)0, sacc, false, false);
      }
      // branchless causal+decay mask: one gs load, 2 VALU muls per element
      const int s = si * 16 + l15;
      const float gf = gs[s];
#pragma unroll
      for (int r = 0; r < 8; ++r) {
        const int t = ti * 16 + r + crow;
        Sp[t * LDSP + s] = sacc[r] * sfac[r] * gf;
      }
    }
    __syncthreads();

    // ---- R = diag(decay^t)*(Q @ fwR) + S' @ V ; 8 tiles, 1 per wave ----
    {
      const int ti = wave >> 2;          // 0..1
      const int vi = wave & 3;           // 0..3
      const int qrow  = (ti * 16 + l15) * LDQ;
      const int srowS = (ti * 16 + l15) * LDSP;
      const int vcol  = vi * 16 + l15;
      v8f racc = (v8f){};
#pragma unroll
      for (int kk = 0; kk < DKDIM; kk += 4) {   // inter-chunk: Q @ fwR
        v2f a = *(const v2f*)&Q[qrow + kk + k2];
        v2f bb;
        bb[0] = fwR[(kk + k2 + 0) * LDFW + vcol];
        bb[1] = fwR[(kk + k2 + 1) * LDFW + vcol];
        racc = __builtin_amdgcn_wmma_f32_16x16x4_f32(false, a, false, bb,
                                                     (short)0, racc, false, false);
      }
#pragma unroll
      for (int r = 0; r < 8; ++r) racc[r] *= dfac[r];
#pragma unroll
      for (int kk = 0; kk < CHK; kk += 4) {     // intra-chunk: S' @ V
        v2f a = *(const v2f*)&Sp[srowS + kk + k2];
        v2f bb;
        bb[0] = Vc[(kk + k2 + 0) * LDV + vcol];
        bb[1] = Vc[(kk + k2 + 1) * LDV + vcol];
        racc = __builtin_amdgcn_wmma_f32_16x16x4_f32(false, a, false, bb,
                                                     (short)0, racc, false, false);
      }
#pragma unroll
      for (int r = 0; r < 8; ++r)
        Rs[(ti * 16 + r + crow) * LDR + vcol] = racc[r];
    }

    // ---- fwW = decay^C * fwR + K'^T @ V ; 16 tiles, 2 per wave ----
    {
      const int dki = wave >> 1;                 // 0..3
      const int dkc = dki * 16 + l15;
#pragma unroll 1
      for (int vv = 0; vv < 2; ++vv) {
        const int vi = (wave & 1) * 2 + vv;      // 0..3
        const int vcol = vi * 16 + l15;
        v8f u = (v8f){};
#pragma unroll
        for (int kk = 0; kk < CHK; kk += 4) {
          const int sb = kk + k2;
          v2f a, bb;
          a[0]  = Kc[(sb + 0) * LDQ + dkc] * wsc[sb + 0];  // A = K'^T
          a[1]  = Kc[(sb + 1) * LDQ + dkc] * wsc[sb + 1];
          bb[0] = Vc[(sb + 0) * LDV + vcol];
          bb[1] = Vc[(sb + 1) * LDV + vcol];
          u = __builtin_amdgcn_wmma_f32_16x16x4_f32(false, a, false, bb,
                                                    (short)0, u, false, false);
        }
        // new state written straight to the other fw buffer (no barrier hold)
#pragma unroll
        for (int r = 0; r < 8; ++r) {
          const int fr = (dki * 16 + r + crow) * LDFW + vcol;
          fwW[fr] = u[r] + dC * fwR[fr];
        }
      }
    }
    __syncthreads();   // Rs + fwW writes complete across all waves

    // store Rs chunk to global with coalesced b128 (2 float4 / thread)
#pragma unroll
    for (int i = 0; i < 2; ++i) {
      const int idx = tid + 256 * i;     // float4 index
      const int t   = idx >> 4;
      const int c4  = (idx & 15) * 4;
      v4f val = *(const v4f*)&Rs[t * LDR + c4];
      float* gp = retr + (((size_t)b * S + c0 + t) * NHEAD + h) * DVDIM + c4;
      *(v4f*)gp = val;
    }
  }
}

// ============================== launcher ====================================
extern "C" void kernel_launch(void* const* d_in, const int* in_sizes, int n_in,
                              void* d_out, int out_size, void* d_ws, size_t ws_size,
                              hipStream_t stream) {
  (void)in_sizes; (void)n_in; (void)out_size; (void)ws_size;
  const float* x       = (const float*)d_in[0];
  const float* w_key   = (const float*)d_in[1];
  const float* w_value = (const float*)d_in[2];
  const float* w_query = (const float*)d_in[3];
  const float* w_gate  = (const float*)d_in[4];
  const float* w_out   = (const float*)d_in[5];
  float* out = (float*)d_out;

  const int B = NBATCH, S = SEQLEN, Dm = DMODEL;
  const int NKV = NHEAD * DKDIM;   // 512

  float* ws   = (float*)d_ws;
  float* qb   = ws;                         const size_t qn = (size_t)B * S * NKV;
  float* k0b  = qb + qn;                    const size_t kn = (size_t)S * NKV;
  float* v0b  = k0b + kn;
  float* g0b  = v0b + kn;                   const size_t gn = (size_t)S * NHEAD;
  float* retr = g0b + gn;

  // projections (bf16 WMMA GEMMs); k/v/gate only need batch 0
  gemm_bf16_xwT<false><<<dim3(NKV / BN, (B * S) / BM), 256, 0, stream>>>(
      x, w_query, nullptr, qb, B * S, NKV, Dm);
  gemm_bf16_xwT<false><<<dim3(NKV / BN, S / BM), 256, 0, stream>>>(
      x, w_key, nullptr, k0b, S, NKV, Dm);
  gemm_bf16_xwT<false><<<dim3(NKV / BN, S / BM), 256, 0, stream>>>(
      x, w_value, nullptr, v0b, S, NKV, Dm);
  gate_kernel<<<(S * NHEAD) / 8, 256, 0, stream>>>(x, w_gate, g0b, S, Dm);

  // chunked linear-attention scan (fp32 WMMA), 32 WGs x 8 waves
  fwm_scan<<<dim3(NHEAD, B), 256, 0, stream>>>(qb, k0b, v0b, g0b, retr, S);

  // out = retr @ w_out^T + x (residual fused into GEMM epilogue)
  gemm_bf16_xwT<true><<<dim3(Dm / BN, (B * S) / BM), 256, 0, stream>>>(
      retr, w_out, x, out, B * S, Dm, NKV);
}
